// VectorQuantizer_1271310319899
// MI455X (gfx1250) — matrix operand order repaired and checked by reading, not compiled, not run on hardware
//
#include <hip/hip_runtime.h>

// VQ-VAE vector quantizer for MI455X (gfx1250, wave32).
//   K1  compress GEMM  [12608,768]x[768,256]  bf16x3 split WMMA (near-f32 accuracy)
//   K3  codebook distances + argmin           bf16 WMMA 16x16x32, TDM LDS staging
//   K4  gather + deterministic loss partials
//   K5  expand GEMM    [12608,256]x[256,768]  bf16x3 split WMMA
// Distance term: argmin_j ||e_j||^2 - 2 z.e  (||z||^2 cancels); loss = 3*mean((zq-zf)^2);
// STE output == zq numerically.
// WMMA scheduling: dependent same-accumulator WMMAs cost ~4 co-exec slots
// (ISA 7.12.1), so all inner loops rotate >=4 independent accumulators.

#define BQ   64
#define NQ   197
#define CIN  768
#define CD   256
#define NE   8192
#define MTOT (BQ * NQ)      // 12608, divisible by 16
#define MT   (MTOT / 16)    // 788 row tiles
#define NT   (NE / 16)      // 512 codebook tiles
#define LROW 528            // 512B row + 16B TDM pad (bank-conflict spreading)

typedef __attribute__((ext_vector_type(16))) __bf16 v16bf;
typedef __attribute__((ext_vector_type(8)))  __bf16 v8bf;
typedef __attribute__((ext_vector_type(8)))  float  v8f;
typedef __attribute__((ext_vector_type(4)))  unsigned int u32x4;
typedef __attribute__((ext_vector_type(8)))  int  i32x8;
typedef __attribute__((ext_vector_type(4)))  int  i32x4;

#if defined(__AMDGCN__) && __has_builtin(__builtin_amdgcn_tensor_load_to_lds) && \
    __has_builtin(__builtin_amdgcn_s_wait_tensorcnt)
#define USE_TDM 1
#else
#define USE_TDM 0
#endif

#if USE_TDM
// One TDM descriptor: 2D tile 256x16 bf16 from embb row j0 into LDS, padded
// 16B per 512B row (D# group1: data_size=2B, pad_enable, interval=128dw, amount=4dw).
// This toolchain's builtin takes 6 args (extra zero int32x8 group before cpol).
static __device__ __forceinline__ void tdm_load_tile(unsigned ldsAddr, unsigned long long ga) {
  u32x4 g0;
  g0[0] = 1u;                                              // count=1 user descriptor
  g0[1] = ldsAddr;                                         // lds_addr (bytes)
  g0[2] = (unsigned)ga;                                    // global_addr[31:0]
  g0[3] = ((unsigned)(ga >> 32) & 0x01FFFFFFu) | 0x80000000u; // addr[56:32] | type=2
  i32x8 g1;
  g1[0] = 0x07910000;        // wg_mask=0 | data_size=1(2B) | pad_en | intv=6(128dw) | amt=3(4dw)
  g1[1] = 0x01000000;        // tensor_dim0 = 256  (bits 63:48)
  g1[2] = 0x20000000;        // tensor_dim1 = 8192 (bits 111:80 low16)
  g1[3] = 0x01000000;        // tile_dim0 = 256    (bits 127:112)
  g1[4] = 16;                // tile_dim1 = 16
  g1[5] = CD;                // tensor_dim0_stride = 256
  g1[6] = 0;
  g1[7] = 0;
  i32x4 gz4 = {0, 0, 0, 0};
  i32x8 gz8 = {0, 0, 0, 0, 0, 0, 0, 0};
  __builtin_amdgcn_tensor_load_to_lds(g0, g1, gz4, gz4, gz8, 0);
}
#endif

// Split x into bf16 head h and bf16 tail (x - h).
static __device__ __forceinline__ void bsplit(float x, __bf16& h, __bf16& l) {
  h = (__bf16)x;
  l = (__bf16)(x - (float)h);
}

// ---------------------------------------------------------------------------
// Weight prep: transpose + bf16x2 split so WMMA B-operands are contiguous.
// Wct[n][k] = split(Wc[k][n])  (256x768);  Wet[n][k] = split(We[k][n]) (768x256)
// ---------------------------------------------------------------------------
__global__ void k_split_wc(const float* __restrict__ Wc,
                           __bf16* __restrict__ Wh, __bf16* __restrict__ Wl) {
  const int i = blockIdx.x * 256 + threadIdx.x;       // over 768*256
  const int k = i / CD, n = i % CD;
  __bf16 h, l;
  bsplit(Wc[i], h, l);
  Wh[(size_t)n * CIN + k] = h;
  Wl[(size_t)n * CIN + k] = l;
}

__global__ void k_split_we(const float* __restrict__ We,
                           __bf16* __restrict__ Wh, __bf16* __restrict__ Wl) {
  const int i = blockIdx.x * 256 + threadIdx.x;       // over 256*768
  const int k = i / CIN, n = i % CIN;
  __bf16 h, l;
  bsplit(We[i], h, l);
  Wh[(size_t)n * CD + k] = h;
  Wl[(size_t)n * CD + k] = l;
}

// ---------------------------------------------------------------------------
// K2: emb -> bf16 copy + ||e||^2 per row (one wave per codebook row).
// ---------------------------------------------------------------------------
__global__ void k_embprep(const float* __restrict__ emb,
                          __bf16* __restrict__ embb, float* __restrict__ enorm) {
  const int lane = threadIdx.x & 31;
  const int wave = threadIdx.x >> 5;
  const int row  = blockIdx.x * 8 + wave;
  const float* e = emb + (size_t)row * CD;
  __bf16* eb = embb + (size_t)row * CD;
  float s = 0.f;
#pragma unroll
  for (int i = 0; i < 8; ++i) {
    const float v = e[lane + 32 * i];
    s += v * v;
    eb[lane + 32 * i] = (__bf16)v;
  }
#pragma unroll
  for (int off = 16; off > 0; off >>= 1) s += __shfl_xor(s, off, 32);
  if (lane == 0) enorm[row] = s;
}

// ---------------------------------------------------------------------------
// K1: zf = z @ Wc + bc via bf16x3 split WMMA. One wave = one M tile x 8 N
// tiles; A split on the fly from f32; B from pre-split transposed weights.
// Inner loop: 4 N-tiles' B in registers, split terms issued term-major so
// consecutive WMMAs always hit different accumulators (hazard hiding).
// ---------------------------------------------------------------------------
__global__ void k_compress(const float* __restrict__ z,
                           const __bf16* __restrict__ Wh, const __bf16* __restrict__ Wl,
                           const float* __restrict__ bc,
                           float* __restrict__ zf, __bf16* __restrict__ zfb) {
  const int lane = threadIdx.x & 31;
  const int wave = threadIdx.x >> 5;
  const int tg = blockIdx.x * 8 + wave;   // mt*2 + g, exact fit
  const int mt = tg >> 1, g = tg & 1;
  const int jn = lane & 15, hi = lane >> 4;
  const int m  = mt * 16 + jn;
  const int nb = g * 128 + jn;
  v8f acc[8] = {};
  const float* zrow = z + (size_t)m * CIN;
  for (int c = 0; c < CIN / 32; ++c) {
    const int kb = c * 32 + hi * 16;
    v8f f0 = *(const v8f*)(zrow + kb);
    v8f f1 = *(const v8f*)(zrow + kb + 8);
    v16bf ah, al;
#pragma unroll
    for (int i = 0; i < 8; ++i) {
      __bf16 h, l;
      bsplit(f0[i], h, l); ah[i] = h;     al[i] = l;
      bsplit(f1[i], h, l); ah[i + 8] = h; al[i + 8] = l;
    }
#pragma unroll
    for (int t0 = 0; t0 < 8; t0 += 4) {
      v16bf bh[4], bl[4];
#pragma unroll
      for (int tt = 0; tt < 4; ++tt) {
        const size_t off = (size_t)(nb + (t0 + tt) * 16) * CIN + c * 32 + hi * 16;
        bh[tt] = *(const v16bf*)(Wh + off);
        bl[tt] = *(const v16bf*)(Wl + off);
      }
#pragma unroll
      for (int tt = 0; tt < 4; ++tt)
        acc[t0 + tt] = __builtin_amdgcn_wmma_f32_16x16x32_bf16(false, ah, false, bh[tt], (short)0, acc[t0 + tt], false, false);
#pragma unroll
      for (int tt = 0; tt < 4; ++tt)
        acc[t0 + tt] = __builtin_amdgcn_wmma_f32_16x16x32_bf16(false, al, false, bh[tt], (short)0, acc[t0 + tt], false, false);
#pragma unroll
      for (int tt = 0; tt < 4; ++tt)
        acc[t0 + tt] = __builtin_amdgcn_wmma_f32_16x16x32_bf16(false, ah, false, bl[tt], (short)0, acc[t0 + tt], false, false);
    }
  }
#pragma unroll
  for (int t = 0; t < 8; ++t) {
    const int n = nb + t * 16;
    const float bias = bc[n];
#pragma unroll
    for (int r = 0; r < 8; ++r) {
      const int row = mt * 16 + r + 8 * hi;
      const float v = acc[t][r] + bias;
      zf[(size_t)row * CD + n]  = v;
      zfb[(size_t)row * CD + n] = (__bf16)v;
    }
  }
}

// ---------------------------------------------------------------------------
// K3: distances + argmin. 8 waves/block each own one 16-row M tile; the
// 16x256 bf16 codebook tile is staged ONCE per block into LDS (TDM double
// buffer; cooperative copy fallback). 4 rotating accumulators hide the
// WMMA->WMMA dependency bubble in the K-chunk chain.
// ---------------------------------------------------------------------------
__global__ void k_argmin(const __bf16* __restrict__ zfb, const __bf16* __restrict__ embb,
                         const float* __restrict__ enorm, int* __restrict__ idxOut) {
  __shared__ __align__(128) char lbuf[2][16 * LROW];
  const int lane = threadIdx.x & 31;
  const int wave = threadIdx.x >> 5;
  const int hi = lane >> 4, jn = lane & 15;
  const int mt  = blockIdx.x * 8 + wave;
  const int mtc = mt < MT ? mt : MT - 1;          // clamp: duplicates are identical
  const int m   = mtc * 16 + jn;

  // A tile (16x256) in register, contiguous k = 16*hi + s per 32-k chunk.
  v16bf a[8];
  const __bf16* arow = zfb + (size_t)m * CD;
#pragma unroll
  for (int c = 0; c < 8; ++c) {
    const int kb = c * 32 + hi * 16;
    v8bf lo = *(const v8bf*)(arow + kb);
    v8bf up = *(const v8bf*)(arow + kb + 8);
    a[c] = __builtin_shufflevector(lo, up, 0,1,2,3,4,5,6,7,8,9,10,11,12,13,14,15);
  }

  float bmin[8]; int bidx[8];
#pragma unroll
  for (int r = 0; r < 8; ++r) { bmin[r] = 3.4e38f; bidx[r] = 0; }

#if USE_TDM
  const unsigned long long gbase = (unsigned long long)(const void*)embb;
  const unsigned lds[2] = { (unsigned)(size_t)(void*)&lbuf[0][0],
                            (unsigned)(size_t)(void*)&lbuf[1][0] };
  if (wave == 0) tdm_load_tile(lds[0], gbase);
#else
  {
    const unsigned* src = (const unsigned*)embb;
    for (int d = threadIdx.x; d < 16 * 128; d += 256)
      *(unsigned*)(&lbuf[0][0] + (d >> 7) * LROW + (d & 127) * 4) = src[d];
  }
#endif

  for (int jt = 0; jt < NT; ++jt) {
#if USE_TDM
    if (wave == 0) {
      if (jt + 1 < NT) {
        tdm_load_tile(lds[(jt + 1) & 1], gbase + (unsigned long long)(jt + 1) * (16 * CD * 2));
        __builtin_amdgcn_s_wait_tensorcnt(1);    // current tile complete
      } else {
        __builtin_amdgcn_s_wait_tensorcnt(0);
      }
    }
#endif
    __syncthreads();                             // tile jt visible to all waves

    // B 32x16 per K-chunk: lane = codebook row jn, 32B contiguous K segment.
    const char* bbase = &lbuf[jt & 1][0] + jn * LROW + hi * 32;
    v8f c4[4] = {};
#pragma unroll
    for (int c = 0; c < 8; ++c) {
      v8bf blo = *(const v8bf*)(bbase + c * 64);
      v8bf bup = *(const v8bf*)(bbase + c * 64 + 16);
      v16bf b = __builtin_shufflevector(blo, bup, 0,1,2,3,4,5,6,7,8,9,10,11,12,13,14,15);
      c4[c & 3] = __builtin_amdgcn_wmma_f32_16x16x32_bf16(false, a[c], false, b, (short)0, c4[c & 3], false, false);
    }
    const int j = jt * 16 + jn;
    const float en = enorm[j];
#pragma unroll
    for (int r = 0; r < 8; ++r) {
      const float dot = (c4[0][r] + c4[1][r]) + (c4[2][r] + c4[3][r]);
      const float d = en - 2.0f * dot;
      const bool better = (d < bmin[r]) || (d == bmin[r] && j < bidx[r]);
      bmin[r] = better ? d : bmin[r];
      bidx[r] = better ? j : bidx[r];
    }
    __syncthreads();                             // reads done before buffer reuse
#if !USE_TDM
    if (jt + 1 < NT) {
      const unsigned* src = (const unsigned*)(embb + (size_t)(jt + 1) * 16 * CD);
      for (int d = threadIdx.x; d < 16 * 128; d += 256)
        *(unsigned*)(&lbuf[(jt + 1) & 1][0] + (d >> 7) * LROW + (d & 127) * 4) = src[d];
    }
#endif
  }

  // reduce the 16 candidate columns per row (xor<16 stays within half-wave)
#pragma unroll
  for (int r = 0; r < 8; ++r) {
#pragma unroll
    for (int off = 8; off > 0; off >>= 1) {
      const float om = __shfl_xor(bmin[r], off, 32);
      const int   oi = __shfl_xor(bidx[r], off, 32);
      const bool better = (om < bmin[r]) || (om == bmin[r] && oi < bidx[r]);
      bmin[r] = better ? om : bmin[r];
      bidx[r] = better ? oi : bidx[r];
    }
  }
  if (mt < MT && (lane & 15) == 0) {
#pragma unroll
    for (int r = 0; r < 8; ++r) idxOut[mt * 16 + r + 8 * hi] = bidx[r];
  }
}

// ---------------------------------------------------------------------------
// K4: z_q = emb[idx] (f32); per-block partial of sum((z_q-zf)^2). Deterministic.
// ---------------------------------------------------------------------------
__global__ void k_gather_loss(const float* __restrict__ emb, const int* __restrict__ idx,
                              const float* __restrict__ zf, float* __restrict__ zq,
                              float* __restrict__ partial) {
  __shared__ float sp[8];
  const int lane = threadIdx.x & 31;
  const int wave = threadIdx.x >> 5;
  const int m    = blockIdx.x * 8 + wave;
  const float* e  = emb + (size_t)idx[m] * CD;
  const float* zr = zf  + (size_t)m * CD;
  float* q        = zq  + (size_t)m * CD;
  float s = 0.f;
#pragma unroll
  for (int i = 0; i < 8; ++i) {
    const float v = e[lane + 32 * i];
    q[lane + 32 * i] = v;
    const float d = v - zr[lane + 32 * i];
    s += d * d;
  }
#pragma unroll
  for (int off = 16; off > 0; off >>= 1) s += __shfl_xor(s, off, 32);
  if (lane == 0) sp[wave] = s;
  __syncthreads();
  if (threadIdx.x == 0) {
    float t = 0.f;
#pragma unroll
    for (int w = 0; w < 8; ++w) t += sp[w];
    partial[blockIdx.x] = t;
  }
}

__global__ void k_loss_final(const float* __restrict__ partial, float* __restrict__ loss) {
  __shared__ float sp[8];
  float s = 0.f;
  for (int i = threadIdx.x; i < MTOT / 8; i += 256) s += partial[i];
#pragma unroll
  for (int off = 16; off > 0; off >>= 1) s += __shfl_xor(s, off, 32);
  if ((threadIdx.x & 31) == 0) sp[threadIdx.x >> 5] = s;
  __syncthreads();
  if (threadIdx.x == 0) {
    float t = 0.f;
#pragma unroll
    for (int w = 0; w < 8; ++w) t += sp[w];
    *loss = t * (3.0f / (float)((size_t)MTOT * CD));
  }
}

// ---------------------------------------------------------------------------
// K5: out = zq @ We + be via bf16x3 split WMMA (same scheme as K1, K=256).
// ---------------------------------------------------------------------------
__global__ void k_expand(const float* __restrict__ zq,
                         const __bf16* __restrict__ Wh, const __bf16* __restrict__ Wl,
                         const float* __restrict__ be, float* __restrict__ out) {
  const int lane = threadIdx.x & 31;
  const int wave = threadIdx.x >> 5;
  const int tg = blockIdx.x * 8 + wave;   // mt*6 + g, exact fit
  const int mt = tg / 6, g = tg % 6;
  const int jn = lane & 15, hi = lane >> 4;
  const int m  = mt * 16 + jn;
  const int nb = g * 128 + jn;
  v8f acc[8] = {};
  const float* zrow = zq + (size_t)m * CD;
  for (int c = 0; c < CD / 32; ++c) {
    const int kb = c * 32 + hi * 16;
    v8f f0 = *(const v8f*)(zrow + kb);
    v8f f1 = *(const v8f*)(zrow + kb + 8);
    v16bf ah, al;
#pragma unroll
    for (int i = 0; i < 8; ++i) {
      __bf16 h, l;
      bsplit(f0[i], h, l); ah[i] = h;     al[i] = l;
      bsplit(f1[i], h, l); ah[i + 8] = h; al[i + 8] = l;
    }
#pragma unroll
    for (int t0 = 0; t0 < 8; t0 += 4) {
      v16bf bh[4], bl[4];
#pragma unroll
      for (int tt = 0; tt < 4; ++tt) {
        const size_t off = (size_t)(nb + (t0 + tt) * 16) * CD + c * 32 + hi * 16;
        bh[tt] = *(const v16bf*)(Wh + off);
        bl[tt] = *(const v16bf*)(Wl + off);
      }
#pragma unroll
      for (int tt = 0; tt < 4; ++tt)
        acc[t0 + tt] = __builtin_amdgcn_wmma_f32_16x16x32_bf16(false, ah, false, bh[tt], (short)0, acc[t0 + tt], false, false);
#pragma unroll
      for (int tt = 0; tt < 4; ++tt)
        acc[t0 + tt] = __builtin_amdgcn_wmma_f32_16x16x32_bf16(false, al, false, bh[tt], (short)0, acc[t0 + tt], false, false);
#pragma unroll
      for (int tt = 0; tt < 4; ++tt)
        acc[t0 + tt] = __builtin_amdgcn_wmma_f32_16x16x32_bf16(false, ah, false, bl[tt], (short)0, acc[t0 + tt], false, false);
    }
  }
#pragma unroll
  for (int t = 0; t < 8; ++t) {
    const int n = nb + t * 16;
    const float bias = be[n];
#pragma unroll
    for (int r = 0; r < 8; ++r) {
      const int row = mt * 16 + r + 8 * hi;
      out[(size_t)row * CIN + n] = acc[t][r] + bias;
    }
  }
}

extern "C" void kernel_launch(void* const* d_in, const int* in_sizes, int n_in,
                              void* d_out, int out_size, void* d_ws, size_t ws_size,
                              hipStream_t stream) {
  const float* z   = (const float*)d_in[0];   // [64,197,768]
  const float* emb = (const float*)d_in[1];   // [8192,256]
  const float* Wc  = (const float*)d_in[2];   // [768,256]
  const float* bc  = (const float*)d_in[3];   // [256]
  const float* We  = (const float*)d_in[4];   // [256,768]
  const float* be  = (const float*)d_in[5];   // [768]

  float* out  = (float*)d_out;                // [M, CIN] flat
  float* loss = out + (size_t)MTOT * CIN;     // final scalar

  char* ws = (char*)d_ws;
  auto salloc = [&](size_t bytes) {
    char* p = ws;
    ws += (bytes + 255) & ~(size_t)255;
    return (void*)p;
  };
  float*  zf      = (float*)salloc(sizeof(float) * (size_t)MTOT * CD);
  float*  zq      = (float*)salloc(sizeof(float) * (size_t)MTOT * CD);
  float*  enorm   = (float*)salloc(sizeof(float) * NE);
  float*  partial = (float*)salloc(sizeof(float) * (MTOT / 8));
  __bf16* zfb     = (__bf16*)salloc(2u * (size_t)MTOT * CD);
  __bf16* embb    = (__bf16*)salloc(2u * (size_t)NE * CD);
  __bf16* WcH     = (__bf16*)salloc(2u * (size_t)CIN * CD);
  __bf16* WcL     = (__bf16*)salloc(2u * (size_t)CIN * CD);
  __bf16* WeH     = (__bf16*)salloc(2u * (size_t)CIN * CD);
  __bf16* WeL     = (__bf16*)salloc(2u * (size_t)CIN * CD);
  int*    idx     = (int*)salloc(sizeof(int) * MTOT);

  k_split_wc   <<<(CIN * CD) / 256, 256, 0, stream>>>(Wc, WcH, WcL);
  k_split_we   <<<(CIN * CD) / 256, 256, 0, stream>>>(We, WeH, WeL);
  k_embprep    <<<NE / 8,           256, 0, stream>>>(emb, embb, enorm);
  k_compress   <<<(MT * 2) / 8,     256, 0, stream>>>(z, WcH, WcL, bc, zf, zfb);
  k_argmin     <<<(MT + 7) / 8,     256, 0, stream>>>(zfb, embb, enorm, idx);
  k_gather_loss<<<MTOT / 8,         256, 0, stream>>>(emb, idx, zf, zq, partial);
  k_loss_final <<<1,                256, 0, stream>>>(partial, loss);
  k_expand     <<<(MT * 6) / 8,     256, 0, stream>>>(zq, WeH, WeL, be, out);
}
